// SliceRenderer_50457275793906
// MI455X (gfx1250) — matrix operand order repaired and checked by reading, not compiled, not run on hardware
//
#include <hip/hip_runtime.h>
#include <hip/hip_bf16.h>

typedef __attribute__((ext_vector_type(16))) _Float16 v16h;
typedef __attribute__((ext_vector_type(8)))  float    v8f;
typedef __attribute__((ext_vector_type(4)))  unsigned u32x4;

#define IMG_W   512
#define IMG_H   512
#define TILE    128       // 128x128 output tile per workgroup
#define KT      32        // WMMA K per inner iteration
#define KCHUNK  256       // gaussians per amortized param stage (8 tiles)
#define ZSPLIT  8         // K-split factor (gridDim.z)
#define A_PITCH 34        // halfs per A row
#define B_PITCH 136       // halfs per B row (16B multiple)
#define EPSF    1e-8f
#define HLOG2E  0.7213475204444817f   // 0.5*log2(e)
#define SQH     0.8493218002880191f   // sqrt(0.5*log2(e))
#define NEG_BIG (-1.0e30f)

__global__ void zero_f32(float* __restrict__ p, int n) {
  int i = blockIdx.x * blockDim.x + threadIdx.x;
  if (i < n) p[i] = 0.0f;
}

__global__ __launch_bounds__(256)
void slice_wmma(const float* __restrict__ pos,
                const float* __restrict__ scl,
                const float* __restrict__ opa,
                const float* __restrict__ inten,
                const int*   __restrict__ ztgt,
                float* __restrict__ out, int n)
{
  __shared__ __attribute__((aligned(16))) _Float16 Alds[TILE * A_PITCH];
  __shared__ __attribute__((aligned(16))) _Float16 Blds[KT * B_PITCH];
  // per-gaussian params for one KCHUNK: row-center, col-center, scaled
  // inverse sigmas, and log2-domain effective weight
  __shared__ float sPx[KCHUNK], sPy[KCHUNK], sIsy[KCHUNK], sIsx[KCHUNK], sLef[KCHUNK];

  const int tid  = threadIdx.x;
  const int lane = tid & 31;
  const int wv   = tid >> 5;

  const int rowBase = blockIdx.y * TILE;
  const int colBase = blockIdx.x * TILE;
  const int wRow = (wv & 3) * 32;      // 4x2 wave grid: 32 rows x 64 cols each
  const int wCol = (wv >> 2) * 64;

  v8f acc[2][4];
#pragma unroll
  for (int mi = 0; mi < 2; ++mi)
#pragma unroll
    for (int ni = 0; ni < 4; ++ni)
      acc[mi][ni] = (v8f){0.f,0.f,0.f,0.f,0.f,0.f,0.f,0.f};

  const float zt = (float)ztgt[0];

  const int kPer = ((n + ZSPLIT * KT - 1) / (ZSPLIT * KT)) * KT;
  const int kBeg = blockIdx.z * kPer;
  int kStop = kBeg + kPer;
  if (kStop > n) kStop = n;

  // fill mappings
  const int a_kp = tid & 15;    // A: k-pair index (k = 2*a_kp, 2*a_kp+1)
  const int a_rg = tid >> 4;    // A: row group (8 rows)
  const int b_k  = tid & 31;    // B: k row
  const int b_cg = tid >> 5;    // B: col group (16 cols)

  // WMMA A-fragment addressing (16-bit A 16x32 layout, ISA 7.12.2)
  const int mloc  = lane & 15;
  const int khalf = (lane >> 4) << 3;

  for (int kb = kBeg; kb < kStop; kb += KCHUNK) {
    // ---- amortized param stage: 1 gaussian per thread ----
    {
      int g = kb + tid;
      float px = 0.f, py = 0.f, isy = 0.f, isx = 0.f, lef = NEG_BIG;
      if (g < n) {
        px = pos[g * 3 + 0];
        py = pos[g * 3 + 1];
        float pz = pos[g * 3 + 2];
        float zd = (zt - pz) * __builtin_amdgcn_rcpf(scl[g * 3 + 2] + EPSF);
        float l_oi = __builtin_amdgcn_logf(opa[g] * inten[g]);     // log2
        lef = (__builtin_fabsf(zd) < 3.0f) ? fmaf(-zd * zd, HLOG2E, l_oi)
                                           : NEG_BIG;
        isy = __builtin_amdgcn_rcpf(scl[g * 3 + 0] + EPSF) * SQH;
        isx = __builtin_amdgcn_rcpf(scl[g * 3 + 1] + EPSF) * SQH;
      }
      sPx[tid] = px; sPy[tid] = py;
      sIsy[tid] = isy; sIsx[tid] = isx; sLef[tid] = lef;
    }
    __syncthreads();

    for (int t = 0; t < KCHUNK / KT; ++t) {
      int k0 = kb + t * KT;
      if (k0 >= kStop) break;
      const int kpar = t * KT;

      // ---- A fill: 2 gaussians x 8 rows, packed b32 stores ----
      {
        int i0 = kpar + 2 * a_kp;
        float px0 = sPx[i0],  px1 = sPx[i0 + 1];
        float is0 = sIsy[i0], is1 = sIsy[i0 + 1];
        float l0  = sLef[i0], l1  = sLef[i0 + 1];
#pragma unroll
        for (int j = 0; j < 8; ++j) {
          int r = a_rg * 8 + j;
          float fy = (float)(rowBase + r);
          float d0 = (fy - px0) * is0;
          float d1 = (fy - px1) * is1;
          union { _Float16 h[2]; unsigned u; } pk;
          pk.h[0] = (_Float16)__builtin_amdgcn_exp2f(fmaf(-d0, d0, l0));
          pk.h[1] = (_Float16)__builtin_amdgcn_exp2f(fmaf(-d1, d1, l1));
          *(unsigned*)&Alds[r * A_PITCH + 2 * a_kp] = pk.u;
        }
      }
      // ---- B fill: 1 gaussian x 16 cols, two b128 stores ----
      {
        int i0 = kpar + b_k;
        float py = sPy[i0], is = sIsx[i0];
        union { _Float16 h[16]; u32x4 q[2]; } pk;
#pragma unroll
        for (int j = 0; j < 16; ++j) {
          float fx = (float)(colBase + b_cg * 16 + j);
          float d = (fx - py) * is;
          pk.h[j] = (_Float16)__builtin_amdgcn_exp2f(-(d * d));
        }
        u32x4* dst = (u32x4*)&Blds[b_k * B_PITCH + b_cg * 16];
        dst[0] = pk.q[0];
        dst[1] = pk.q[1];
      }
      __syncthreads();

      // ---- fragment loads ----
      union { v16h h; unsigned u[8]; } af[2];
#pragma unroll
      for (int mi = 0; mi < 2; ++mi) {
        const _Float16* arow = &Alds[(wRow + mi * 16 + mloc) * A_PITCH];
#pragma unroll
        for (int j = 0; j < 8; ++j) {
          int kk = khalf + ((j < 4) ? (2 * j) : (16 + 2 * (j - 4)));
          af[mi].u[j] = *(const unsigned*)(arow + kk);
        }
      }
      union { v16h h; u32x4 q[2]; } bf[4];
#pragma unroll
      for (int ni = 0; ni < 4; ++ni) {
        const _Float16* bcol = &Blds[wCol + ni * 16 + mloc];
#pragma unroll
        for (int s = 0; s < 16; ++s) {
          int kk = khalf + ((s < 8) ? s : (8 + s));
          bf[ni].h[s] = bcol[kk * B_PITCH];
        }
      }

      // ---- 8 WMMAs ----
#pragma unroll
      for (int mi = 0; mi < 2; ++mi)
#pragma unroll
        for (int ni = 0; ni < 4; ++ni)
          acc[mi][ni] = __builtin_amdgcn_wmma_f32_16x16x32_f16(
              false, af[mi].h, false, bf[ni].h,
              (short)0, acc[mi][ni], false, false);

      __syncthreads();   // frag loads done before next fill overwrites LDS
    }
  }

  // ---- epilogue: C/D layout VGPR i -> M = i (+8 lanes 16-31), N = lane%16
  const int nOut = lane & 15;
  const int mAdd = (lane >> 4) * 8;
#pragma unroll
  for (int mi = 0; mi < 2; ++mi)
#pragma unroll
    for (int ni = 0; ni < 4; ++ni)
#pragma unroll
      for (int i = 0; i < 8; ++i) {
        int m = rowBase + wRow + mi * 16 + mAdd + i;
        int c = colBase + wCol + ni * 16 + nOut;
        __hip_atomic_fetch_add(&out[m * IMG_W + c], acc[mi][ni][i],
                               __ATOMIC_RELAXED, __HIP_MEMORY_SCOPE_AGENT);
      }
}

extern "C" void kernel_launch(void* const* d_in, const int* in_sizes, int n_in,
                              void* d_out, int out_size, void* d_ws, size_t ws_size,
                              hipStream_t stream) {
  const float* pos   = (const float*)d_in[0];
  const float* scl   = (const float*)d_in[1];
  const float* opa   = (const float*)d_in[2];
  const float* inten = (const float*)d_in[3];
  const int*   ztgt  = (const int*)d_in[4];
  float* out = (float*)d_out;
  const int n = in_sizes[2];   // N gaussians

  zero_f32<<<(out_size + 255) / 256, 256, 0, stream>>>(out, out_size);

  dim3 grid(IMG_W / TILE, IMG_H / TILE, ZSPLIT);   // 4 x 4 x 8 = 128 WGs
  slice_wmma<<<grid, 256, 0, stream>>>(pos, scl, opa, inten, ztgt, out, n);
}
